// UTSPGNN_31533649887299
// MI455X (gfx1250) — matrix operand order repaired and checked by reading, not compiled, not run on hardware
//
#include <hip/hip_runtime.h>
#include <hip/hip_fp16.h>

typedef __attribute__((ext_vector_type(2)))  float    v2f;
typedef __attribute__((ext_vector_type(8)))  float    v8f;
typedef __attribute__((ext_vector_type(8)))  _Float16 v8h;
typedef __attribute__((ext_vector_type(16))) _Float16 v16h;

#define HID   128
#define NPTS  1024
#define NB    4
#define NH    8
#define DH    16
#define NLAY  3
#define ALPHA 0.2f

__device__ __forceinline__ float lrelu(float x) { return x > 0.f ? x : ALPHA * x; }

// ---------------------------------------------------------------------------
// h0 = coords @ emb_W + emb_b     (4096 rows x 128)
// ---------------------------------------------------------------------------
__global__ __launch_bounds__(256) void embed_kernel(
    const float* __restrict__ coords, const float* __restrict__ emb_W,
    const float* __restrict__ emb_b, float* __restrict__ h) {
  int tid = blockIdx.x * 256 + threadIdx.x;        // B*N*HID = 524288
  int row = tid >> 7, c = tid & 127;
  float x = coords[row * 2 + 0], y = coords[row * 2 + 1];
  h[tid] = x * emb_W[c] + y * emb_W[HID + c] + emb_b[c];
}

// ---------------------------------------------------------------------------
// Wh = h @ W   via v_wmma_f32_16x16x4_f32 (K=128 -> 32 WMMAs per 16x16 tile).
// Also writes transposed f16 copy WhT[b][hid][n] for the attention B-operand.
// One wave per 16x16 output tile; 4096x128 -> 2048 waves.
// ---------------------------------------------------------------------------
__global__ __launch_bounds__(256) void wh_kernel(
    const float* __restrict__ h, const float* __restrict__ W,
    float* __restrict__ Wh, _Float16* __restrict__ WhT) {
  int wave  = (blockIdx.x * blockDim.x + threadIdx.x) >> 5;
  int lane  = threadIdx.x & 31;
  int itile = wave >> 3;              // 0..255 (16-row tiles over B*N)
  int jtile = wave & 7;               // 0..7   (16-col tiles over HID)
  int hi = lane >> 4, lm = lane & 15;
  int arow = itile * 16 + lm;         // A-matrix row for this lane
  int col  = jtile * 16 + lm;         // B/C column for this lane
  v8f c = {};
  #pragma unroll 4
  for (int k0 = 0; k0 < HID; k0 += 4) {
    // A 16x4 f32: lane<16 holds K={k0,k0+1}, lane>=16 holds K={k0+2,k0+3}
    v2f a = *(const v2f*)(h + arow * HID + k0 + 2 * hi);
    // B 4x16 f32: same K split across lane halves, column = lm
    v2f b;
    b.x = W[(k0 + 2 * hi) * HID + col];
    b.y = W[(k0 + 2 * hi + 1) * HID + col];
    c = __builtin_amdgcn_wmma_f32_16x16x4_f32(false, a, false, b, (short)0, c,
                                              false, false);
  }
  // C layout: lane L, VGPR v -> row v + 8*(L/16), col lm
  int n0 = itile * 16 + 8 * hi;
  v8h tv;
  #pragma unroll
  for (int v = 0; v < 8; ++v) {
    int grow = n0 + v;
    Wh[grow * HID + col] = c[v];
    tv[v] = (_Float16)c[v];
  }
  int bb = n0 >> 10;                               // batch index (tiles never straddle)
  *(v8h*)(WhT + (size_t)(bb * HID + col) * NPTS + (n0 & 1023)) = tv;
}

// ---------------------------------------------------------------------------
// s_src/s_dst[b,h,n] = sum_d Wh[b,n,h,d] * a_{src,dst}[h,d]
// ---------------------------------------------------------------------------
__global__ __launch_bounds__(256) void score_kernel(
    const float* __restrict__ Wh, const float* __restrict__ a,
    float* __restrict__ s_src, float* __restrict__ s_dst) {
  int tid = blockIdx.x * 256 + threadIdx.x;        // B*H*N = 32768
  int n = tid & 1023;
  int bh = tid >> 10;
  int b = bh >> 3, hh = bh & 7;
  const float* w = Wh + (size_t)(b * NPTS + n) * HID + hh * DH;
  float ss = 0.f, sd = 0.f;
  #pragma unroll
  for (int d = 0; d < DH; ++d) {
    float x = w[d];
    ss += x * a[hh * 2 * DH + d];
    sd += x * a[hh * 2 * DH + DH + d];
  }
  s_src[bh * NPTS + n] = ss;
  s_dst[bh * NPTS + n] = sd;
}

// max_j s_dst per (b,h): valid softmax shift since leaky_relu is monotone.
__global__ __launch_bounds__(256) void smax_kernel(
    const float* __restrict__ s_dst, float* __restrict__ smax) {
  __shared__ float red[256];
  int bh = blockIdx.x, t = threadIdx.x;
  float m = -3.4e38f;
  for (int n = t; n < NPTS; n += 256) m = fmaxf(m, s_dst[bh * NPTS + n]);
  red[t] = m;
  __syncthreads();
  for (int s = 128; s > 0; s >>= 1) {
    if (t < s) red[t] = fmaxf(red[t], red[t + s]);
    __syncthreads();
  }
  if (t == 0) smax[bh] = red[0];
}

// ---------------------------------------------------------------------------
// Fused softmax + aggregation: out[b,i,h,:] = softmax_j(e_ij) @ Wh[b,:,h,:]
// One wave per (b,h, 16-row i-tile) -> 2048 waves. K chunks of 32 j's via
// v_wmma_f32_16x16x32_f16 (P tile in f16, Wh^T tile in f16, f32 accum).
// No online rescaling needed: shift m_i = lrelu(s_i + max_j s_j) bounds all e.
// s_j operands are fetched as two aligned 32B vector loads per chunk (L0
// broadcasts across the 16 lanes sharing an address) -- no ds_bpermute in
// the hot loop.
// ---------------------------------------------------------------------------
__global__ __launch_bounds__(256) void attn_kernel(
    const float* __restrict__ s_src, const float* __restrict__ s_dst,
    const float* __restrict__ smax, const _Float16* __restrict__ WhT,
    float* __restrict__ hout, int do_relu) {
  int wave = (blockIdx.x * blockDim.x + threadIdx.x) >> 5;
  int lane = threadIdx.x & 31;
  int bh = wave >> 6, itile = wave & 63;
  int b = bh >> 3, hh = bh & 7;
  int hi = lane >> 4, lm = lane & 15;
  int iglob = itile * 16 + lm;                     // A-matrix row for this lane
  float si = s_src[bh * NPTS + iglob];
  float m = lrelu(si + smax[bh]);
  const float* sdp = s_dst + bh * NPTS;
  // B operand: column d = lm of this head -> row [b][hh*16+lm][*] of WhT
  const _Float16* wb = WhT + (size_t)(b * HID + hh * DH + lm) * NPTS;
  v8f c = {};
  float rsum = 0.f;
  #pragma unroll 2
  for (int j0 = 0; j0 < NPTS; j0 += 32) {
    // A f16 16x32 K split: lane half hi needs K = 8*hi+t and 16+8*hi+t.
    // Those s_j are contiguous: two aligned 32-byte loads.
    v8f sj1v = *(const v8f*)(sdp + j0 + 8 * hi);
    v8f sj2v = *(const v8f*)(sdp + j0 + 16 + 8 * hi);
    v16h a;
    #pragma unroll
    for (int t = 0; t < 8; ++t) {
      int k = 8 * hi + t;
      int k2 = k + 16;
      float p1 = (j0 + k == iglob) ? 0.f : __expf(lrelu(si + sj1v[t]) - m);
      float p2 = (j0 + k2 == iglob) ? 0.f : __expf(lrelu(si + sj2v[t]) - m);
      rsum += p1 + p2;
      a[t] = (_Float16)p1;
      a[t + 8] = (_Float16)p2;
    }
    // B f16 32x16: lane half hi holds K = 8*hi + 0..7 and 16 + 8*hi + 0..7
    v8h blo = *(const v8h*)(wb + j0 + 8 * hi);
    v8h bhi = *(const v8h*)(wb + j0 + 16 + 8 * hi);
    v16h bm = __builtin_shufflevector(blo, bhi, 0, 1, 2, 3, 4, 5, 6, 7, 8, 9, 10, 11, 12, 13, 14, 15);
    c = __builtin_amdgcn_wmma_f32_16x16x32_f16(false, a, false, bm, (short)0, c,
                                               false, false);
  }
  // combine the two lane-halves' partial row sums (rows indexed by lm)
  rsum += __shfl_xor(rsum, 16, 32);
  float* outp = hout + (size_t)(b * NPTS + itile * 16) * HID + hh * DH + lm;
  #pragma unroll
  for (int v = 0; v < 8; ++v) {
    int r = 8 * hi + v;                            // C row for this VGPR
    float rs = __shfl(rsum, r, 32);
    float val = c[v] / rs;
    if (do_relu) val = fmaxf(val, 0.f);
    outp[r * HID] = val;
  }
}

// ---------------------------------------------------------------------------
// u = h @ w1, v = h @ w2   (4096 x 128 GEMVs)
// ---------------------------------------------------------------------------
__global__ __launch_bounds__(256) void gemv_kernel(
    const float* __restrict__ h, const float* __restrict__ out_W,
    float* __restrict__ u, float* __restrict__ v) {
  int row = blockIdx.x * 256 + threadIdx.x;        // 4096
  const float* hr = h + (size_t)row * HID;
  float a = 0.f, bb = 0.f;
  #pragma unroll 8
  for (int k = 0; k < HID; ++k) {
    float x = hr[k];
    a += x * out_W[k];
    bb += x * out_W[HID + k];
  }
  u[row] = a;
  v[row] = bb;
}

// heatmap[b,i,j] = sigmoid(u[b,i] + v[b,j] + out_b) * (i != j)
__global__ __launch_bounds__(256) void heatmap_kernel(
    const float* __restrict__ u, const float* __restrict__ v,
    const float* __restrict__ out_b, float* __restrict__ out) {
  int tid = blockIdx.x * 256 + threadIdx.x;        // B*N*N = 4194304
  int j = tid & 1023;
  int bi = tid >> 10;
  int i = bi & 1023;
  int b = bi >> 10;
  float logit = u[b * NPTS + i] + v[b * NPTS + j] + out_b[0];
  float s = 1.f / (1.f + __expf(-logit));
  out[tid] = (i == j) ? 0.f : s;
}

// ---------------------------------------------------------------------------
extern "C" void kernel_launch(void* const* d_in, const int* in_sizes, int n_in,
                              void* d_out, int out_size, void* d_ws,
                              size_t ws_size, hipStream_t stream) {
  const float* coords = (const float*)d_in[0];
  const float* emb_W  = (const float*)d_in[1];
  const float* emb_b  = (const float*)d_in[2];
  const float* gat_W  = (const float*)d_in[3];   // (L,128,128)
  const float* gat_a  = (const float*)d_in[4];   // (L,8,32,1)
  const float* out_W  = (const float*)d_in[5];   // (256,1)
  const float* out_b  = (const float*)d_in[6];

  char* ws = (char*)d_ws;
  size_t off = 0;
  auto alloc = [&](size_t bytes) -> void* {
    void* p = ws + off;
    off = (off + bytes + 255) & ~(size_t)255;
    return p;
  };
  const size_t NROWS = (size_t)NB * NPTS;               // 4096
  float*    hA    = (float*)alloc(NROWS * HID * 4);     // 2 MB
  float*    hB    = (float*)alloc(NROWS * HID * 4);     // 2 MB
  float*    Wh    = (float*)alloc(NROWS * HID * 4);     // 2 MB
  _Float16* WhT   = (_Float16*)alloc((size_t)NB * HID * NPTS * 2);  // 1 MB
  float*    s_src = (float*)alloc((size_t)NB * NH * NPTS * 4);
  float*    s_dst = (float*)alloc((size_t)NB * NH * NPTS * 4);
  float*    smax  = (float*)alloc(NB * NH * 4);
  float*    uu    = (float*)alloc(NROWS * 4);
  float*    vv    = (float*)alloc(NROWS * 4);
  (void)ws_size; (void)in_sizes; (void)n_in; (void)out_size;

  embed_kernel<<<2048, 256, 0, stream>>>(coords, emb_W, emb_b, hA);

  float* hin = hA;
  float* hout = hB;
  for (int l = 0; l < NLAY; ++l) {
    wh_kernel<<<256, 256, 0, stream>>>(hin, gat_W + (size_t)l * HID * HID, Wh, WhT);
    score_kernel<<<128, 256, 0, stream>>>(Wh, gat_a + (size_t)l * NH * 2 * DH,
                                          s_src, s_dst);
    smax_kernel<<<NB * NH, 256, 0, stream>>>(s_dst, smax);
    attn_kernel<<<256, 256, 0, stream>>>(s_src, s_dst, smax, WhT, hout,
                                         l < NLAY - 1 ? 1 : 0);
    float* t = hin; hin = hout; hout = t;              // result now in hin
  }

  gemv_kernel<<<16, 256, 0, stream>>>(hin, out_W, uu, vv);
  heatmap_kernel<<<16384, 256, 0, stream>>>(uu, vv, out_b, (float*)d_out);
}
